// SSPPathIntegrationModel_11261404250906
// MI455X (gfx1250) — matrix operand order repaired and checked by reading, not compiled, not run on hardware
//
#include <hip/hip_runtime.h>
#include <math.h>

#define T_STEPS 100
#define BATCH   2048
#define HID     128
#define ORD     6
#define NIN     2
#define LIND    512
#define SPD     512
#define ROWS_PB 32

typedef __attribute__((ext_vector_type(2)))  float  v2f;
typedef __attribute__((ext_vector_type(8)))  float  v8f;
typedef __attribute__((ext_vector_type(2)))  __bf16 v2bf;
typedef __attribute__((ext_vector_type(8)))  __bf16 v8bf;
typedef __attribute__((ext_vector_type(16))) __bf16 v16bf;

// K-pair-major pitches (in floats) for the scan kernel's LDS tiles:
// element (k, n) lives at [(k>>1)*PITCH + n*2 + (k&1)], so the (kk, kk+1)
// pair a WMMA lane needs is one aligned ds_load_b64.
#define HKT_PITCH 288     // 128 cols * 2 + 32 pad (disjoint banks for lane halves)
#define HT_PITCH  32      // 16 rows * 2

// ---------------------------------------------------------------------------
// Pack a row-major weight W[N][K] (f32) into bf16 WMMA B-fragment order for
// v_wmma_f32_16x16x32_bf16 (B 32x16, contiguous K chosen here to match A: lanes 0-15 K=0-15, lanes 16-31 K=16-31).
// Fragment-major: pack[((ks*NT + nt)*32 + lane)*16 + p]
// ---------------------------------------------------------------------------
__global__ void pack_b_kernel(const float* __restrict__ W, __bf16* __restrict__ pack,
                              int Ndim, int Kdim) {
    const int NT = Ndim >> 4;
    const int KT = Kdim >> 5;
    const int total = KT * NT * 32;
    for (int idx = blockIdx.x * blockDim.x + threadIdx.x; idx < total;
         idx += gridDim.x * blockDim.x) {
        const int lane = idx & 31;
        const int f    = idx >> 5;
        const int nt   = f % NT;
        const int ks   = f / NT;
        const int N    = nt * 16 + (lane & 15);
        const int kb   = ks * 32 + ((lane >> 4) << 4);
        __bf16* dst = pack + (size_t)idx * 16;
        const float* src = W + (size_t)N * Kdim + kb;
        #pragma unroll
        for (int p = 0; p < 16; ++p) dst[p] = (__bf16)src[p];
    }
}

// ---------------------------------------------------------------------------
// LMU recurrence.  One block = 16 batch rows, 8 waves, 100 sequential steps.
// h @ HK^T with v_wmma_f32_16x16x4_f32: one 16-col N-tile per wave, 32-deep
// K-chain; A/B fragments are single ds_load_b64 from K-pair-major LDS tiles.
// ---------------------------------------------------------------------------
__global__ void lmu_scan_kernel(const float* __restrict__ vel,
                                const float* __restrict__ ssp0,
                                const float* __restrict__ IE,
                                const float* __restrict__ HE,
                                const float* __restrict__ ME,
                                const float* __restrict__ IK,
                                const float* __restrict__ HK,
                                const float* __restrict__ MK,
                                const float* __restrict__ AT,
                                const float* __restrict__ BT,
                                const float* __restrict__ WC,
                                const float* __restrict__ WH,
                                __bf16* __restrict__ Hbf) {
    extern __shared__ float smem[];
    float* HKTp = smem;                     // [64][HKT_PITCH]
    float* hTp  = HKTp + 64 * HKT_PITCH;    // [64][HT_PITCH]
    float* RED  = hTp + 64 * HT_PITCH;      // [16][17] u-dot partials
    float* IKs  = RED + 16 * 17;            // [128][2]
    float* MKs  = IKs + 256;                // [128][6]
    float* HEs  = MKs + 768;                // [128]
    float* MEs  = HEs + 128;                // [8]
    float* IEs  = MEs + 8;                  // [4]
    float* ATs  = IEs + 4;                  // [40]
    float* BTs  = ATs + 40;                 // [8]
    float* MS   = BTs + 8;                  // [2][16][8] m double buffer
    float* VTs  = MS + 256;                 // [16][2]

    const int tid  = threadIdx.x;
    const int lane = tid & 31;
    const int wave = tid >> 5;
    const int row0 = blockIdx.x * 16;

    // ---- stage weights ----
    for (int idx = tid; idx < HID * HID; idx += 256) {
        const int n = idx >> 7, k = idx & 127;
        HKTp[(k >> 1) * HKT_PITCH + n * 2 + (k & 1)] = HK[idx];
    }
    for (int idx = tid; idx < HID * NIN; idx += 256) IKs[idx] = IK[idx];
    for (int idx = tid; idx < HID * ORD; idx += 256) MKs[idx] = MK[idx];
    if (tid < HID)       HEs[tid] = HE[tid];
    if (tid < ORD)       MEs[tid] = ME[tid];
    if (tid < NIN)       IEs[tid] = IE[tid];
    if (tid < ORD * ORD) ATs[tid] = AT[tid];
    if (tid < ORD)       BTs[tid] = BT[tid];

    // ---- initial states: h0 = ssp0 @ WH^T, m0 = ssp0 @ WC^T ----
    #pragma unroll
    for (int i = 0; i < 8; ++i) {
        const int o = tid + 256 * i;
        const int r = o & 15, n = o >> 4;
        const float* srow = ssp0 + (size_t)(row0 + r) * SPD;
        const float* wrow = WH + (size_t)n * SPD;
        float acc = 0.f;
        for (int k = 0; k < SPD; ++k) acc += srow[k] * wrow[k];
        hTp[(n >> 1) * HT_PITCH + r * 2 + (n & 1)] = acc;
    }
    if (tid < 16 * ORD) {
        const int r = tid & 15, j = tid >> 4;
        const float* srow = ssp0 + (size_t)(row0 + r) * SPD;
        const float* wrow = WC + (size_t)j * SPD;
        float acc = 0.f;
        for (int k = 0; k < SPD; ++k) acc += srow[k] * wrow[k];
        MS[r * 8 + j] = acc;
    }
    __syncthreads();

    int cur = 0;
    const int n0    = wave * 16;
    const int rA    = lane & 15;
    const int hiSel = (lane >> 4) << 1;    // lanes 16-31 hold K+2 in f32 frags
    const int baseM = (lane >> 4) << 3;
    const int ur    = tid >> 4;            // u-dot: row handled by this thread
    const int us    = tid & 15;            // u-dot: 8-element segment index
    const float* aBase = hTp  + (hiSel >> 1) * HT_PITCH  + rA * 2;
    const float* bBase = HKTp + (hiSel >> 1) * HKT_PITCH + (n0 + rA) * 2;

    for (int t = 0; t < T_STEPS; ++t) {
        if (tid < 32) {
            const int r = tid >> 1, c = tid & 1;
            VTs[r * 2 + c] = vel[((size_t)t * BATCH + row0 + r) * NIN + c];
        }
        if (tid == 0 && t + 1 < T_STEPS)
            __builtin_prefetch(&vel[((size_t)(t + 1) * BATCH + row0) * NIN], 0, 1);

        // u-dot partials: 16 threads per row, 8 elements each
        {
            float p = 0.f;
            const float* hp = hTp + ur * 2;
            #pragma unroll
            for (int kk = 0; kk < 8; ++kk) {
                const int k = us * 8 + kk;
                p += hp[(k >> 1) * HT_PITCH + (k & 1)] * HEs[k];
            }
            RED[ur * 17 + us] = p;
        }
        __syncthreads();                                       // (a)

        // u and m update (one thread per batch row)
        if (tid < 16) {
            const int r = tid;
            float uu = VTs[r * 2 + 0] * IEs[0] + VTs[r * 2 + 1] * IEs[1];
            #pragma unroll
            for (int s = 0; s < 16; ++s) uu += RED[r * 17 + s];
            const float* mr = MS + cur * 128 + r * 8;
            float*       mn = MS + (1 - cur) * 128 + r * 8;
            #pragma unroll
            for (int j = 0; j < ORD; ++j) uu += mr[j] * MEs[j];
            #pragma unroll
            for (int i = 0; i < ORD; ++i) {
                float s = mr[i] + uu * BTs[i];
                #pragma unroll
                for (int j = 0; j < ORD; ++j) s += mr[j] * ATs[i * ORD + j];
                mn[i] = s;
            }
        }
        __syncthreads();                                       // (b)

        // h @ HK^T via fp32 WMMA chain over K=128 (b64 fragment loads)
        v8f acc = {0.f, 0.f, 0.f, 0.f, 0.f, 0.f, 0.f, 0.f};
        #pragma unroll 8
        for (int k0 = 0; k0 < HID; k0 += 4) {
            const v2f a = *(const v2f*)(aBase + (k0 >> 1) * HT_PITCH);
            const v2f b = *(const v2f*)(bBase + (k0 >> 1) * HKT_PITCH);
            acc = __builtin_amdgcn_wmma_f32_16x16x4_f32(false, a, false, b,
                                                        (short)0, acc, false, false);
        }
        __syncthreads();                                       // (c) hTp reads done

        // epilogue: add v@IK^T and m_new@MK^T, tanh, write h back to LDS
        const int N = n0 + rA;
        const float ik0 = IKs[N * 2 + 0], ik1 = IKs[N * 2 + 1];
        #pragma unroll
        for (int i = 0; i < 8; ++i) {
            const int M = baseM + i;
            const float* mn = MS + (1 - cur) * 128 + M * 8;
            float x = acc[i] + ik0 * VTs[M * 2 + 0] + ik1 * VTs[M * 2 + 1];
            #pragma unroll
            for (int j = 0; j < ORD; ++j) x += MKs[N * ORD + j] * mn[j];
            x = tanhf(x);
            hTp[(N >> 1) * HT_PITCH + M * 2 + (N & 1)] = x;
        }
        __syncthreads();                                       // (d) hTp complete

        // coalesced bf16 trace store: pairs of adjacent N packed into 4B
        for (int idx = tid; idx < 16 * 64; idx += 256) {
            const int r  = idx >> 6;
            const int np = idx & 63;
            const v2f hv = *(const v2f*)(hTp + np * HT_PITCH + r * 2);
            v2bf pk;
            pk.x = (__bf16)hv.x;
            pk.y = (__bf16)hv.y;
            *(v2bf*)(Hbf + ((size_t)t * BATCH + row0 + r) * HID + 2 * np) = pk;
        }
        cur ^= 1;
    }
}

// ---------------------------------------------------------------------------
// Fused output projections: 32 rows per block.
//   feat = hid @ Wlin^T + b_lin   (K=128, bf16 WMMA, staged in LDS)
//   out  = feat @ Wssp^T + b_ssp  (K=512, bf16 WMMA)
// ---------------------------------------------------------------------------
__device__ __forceinline__ v16bf load_a_frag(const __bf16* rowBase, int ks, int sel) {
    const __bf16* p = rowBase + ks * 32 + sel;
    v8bf lo = *(const v8bf*)p;
    v8bf hi = *(const v8bf*)(p + 8);
    return __builtin_shufflevector(lo, hi, 0, 1, 2, 3, 4, 5, 6, 7,
                                           8, 9, 10, 11, 12, 13, 14, 15);
}

__global__ void proj_kernel(const __bf16* __restrict__ Hbf,
                            const __bf16* __restrict__ WlinP,
                            const __bf16* __restrict__ WsspP,
                            const float* __restrict__ b_lin,
                            const float* __restrict__ b_ssp,
                            float* __restrict__ out) {
    __shared__ __bf16 hid[ROWS_PB][136];
    __shared__ __bf16 feat[ROWS_PB][520];

    const int tid  = threadIdx.x;
    const int lane = tid & 31;
    const int wave = tid >> 5;
    const size_t rowBase = (size_t)blockIdx.x * ROWS_PB;

    for (int idx = tid; idx < ROWS_PB * HID / 8; idx += 256) {
        const int r  = idx >> 4;
        const int kc = (idx & 15) << 3;
        *(v8bf*)(&hid[r][kc]) = *(const v8bf*)(Hbf + (rowBase + r) * HID + kc);
    }
    __syncthreads();

    const int mloc  = lane & 15;
    const int sel   = (lane >> 4) << 4;
    const int baseM = (lane >> 4) << 3;

    // ---- GEMM1: feat = hid @ Wlin^T + b_lin ----
    {
        v8f acc[2][4] = {};
        #pragma unroll
        for (int ks = 0; ks < HID / 32; ++ks) {
            v16bf a[2];
            #pragma unroll
            for (int mt = 0; mt < 2; ++mt)
                a[mt] = load_a_frag(&hid[mt * 16 + mloc][0], ks, sel);
            #pragma unroll
            for (int q = 0; q < 4; ++q) {
                const int nt = wave * 4 + q;
                const v16bf b = *(const v16bf*)(WlinP +
                    ((size_t)(ks * (LIND / 16) + nt) * 32 + lane) * 16);
                #pragma unroll
                for (int mt = 0; mt < 2; ++mt)
                    acc[mt][q] = __builtin_amdgcn_wmma_f32_16x16x32_bf16(
                        false, a[mt], false, b, (short)0, acc[mt][q], false, false);
            }
        }
        #pragma unroll
        for (int q = 0; q < 4; ++q) {
            const int N = (wave * 4 + q) * 16 + mloc;
            const float bl = b_lin[N];
            #pragma unroll
            for (int mt = 0; mt < 2; ++mt)
                #pragma unroll
                for (int i = 0; i < 8; ++i)
                    feat[mt * 16 + baseM + i][N] = (__bf16)(acc[mt][q][i] + bl);
        }
    }
    __syncthreads();

    // ---- GEMM2: out = feat @ Wssp^T + b_ssp ----
    {
        v8f acc[2][4] = {};
        #pragma unroll
        for (int ks = 0; ks < LIND / 32; ++ks) {
            v16bf a[2];
            #pragma unroll
            for (int mt = 0; mt < 2; ++mt)
                a[mt] = load_a_frag(&feat[mt * 16 + mloc][0], ks, sel);
            #pragma unroll
            for (int q = 0; q < 4; ++q) {
                const int nt = wave * 4 + q;
                const v16bf b = *(const v16bf*)(WsspP +
                    ((size_t)(ks * (SPD / 16) + nt) * 32 + lane) * 16);
                #pragma unroll
                for (int mt = 0; mt < 2; ++mt)
                    acc[mt][q] = __builtin_amdgcn_wmma_f32_16x16x32_bf16(
                        false, a[mt], false, b, (short)0, acc[mt][q], false, false);
            }
        }
        #pragma unroll
        for (int q = 0; q < 4; ++q) {
            const int N = (wave * 4 + q) * 16 + mloc;
            const float bs = b_ssp[N];
            #pragma unroll
            for (int mt = 0; mt < 2; ++mt)
                #pragma unroll
                for (int i = 0; i < 8; ++i)
                    out[(rowBase + mt * 16 + baseM + i) * SPD + N] = acc[mt][q][i] + bs;
        }
    }
}

extern "C" void kernel_launch(void* const* d_in, const int* in_sizes, int n_in,
                              void* d_out, int out_size, void* d_ws, size_t ws_size,
                              hipStream_t stream) {
    const float* vel  = (const float*)d_in[0];
    const float* ssp0 = (const float*)d_in[1];
    const float* IE   = (const float*)d_in[2];
    const float* HE   = (const float*)d_in[3];
    const float* ME   = (const float*)d_in[4];
    const float* IK   = (const float*)d_in[5];
    const float* HK   = (const float*)d_in[6];
    const float* MK   = (const float*)d_in[7];
    const float* AT   = (const float*)d_in[8];
    const float* BT   = (const float*)d_in[9];
    const float* WC   = (const float*)d_in[10];
    const float* WH   = (const float*)d_in[11];
    const float* Wlin = (const float*)d_in[12];
    const float* blin = (const float*)d_in[13];
    const float* Wssp = (const float*)d_in[14];
    const float* bssp = (const float*)d_in[15];
    float* out = (float*)d_out;

    __bf16* Hbf   = (__bf16*)d_ws;
    __bf16* WlinP = Hbf + (size_t)T_STEPS * BATCH * HID;
    __bf16* WsspP = WlinP + (size_t)LIND * HID;

    pack_b_kernel<<<32, 256, 0, stream>>>(Wlin, WlinP, LIND, HID);
    pack_b_kernel<<<128, 256, 0, stream>>>(Wssp, WsspP, SPD, LIND);

    const size_t smem = (size_t)(64 * HKT_PITCH + 64 * HT_PITCH + 16 * 17 + 256 +
                                 768 + 128 + 8 + 4 + 40 + 8 + 256 + 32) * sizeof(float);
    lmu_scan_kernel<<<BATCH / 16, 256, smem, stream>>>(
        vel, ssp0, IE, HE, ME, IK, HK, MK, AT, BT, WC, WH, Hbf);

    proj_kernel<<<(T_STEPS * BATCH) / ROWS_PB, 256, 0, stream>>>(
        Hbf, WlinP, WsspP, blin, bssp, out);
}
